// PatchedQwen3VLMoeTextExperts_26096221290603
// MI455X (gfx1250) — matrix operand order repaired and checked by reading, not compiled, not run on hardware
//
#include <hip/hip_runtime.h>
#include <hip/hip_bf16.h>
#include <math.h>

typedef __bf16 bf16_t;
typedef __attribute__((ext_vector_type(16))) __bf16 v16bf;
typedef __attribute__((ext_vector_type(8)))  __bf16 v8bf;
typedef __attribute__((ext_vector_type(8)))  float  v8f;

constexpr int T_ = 2048;   // tokens
constexpr int K_ = 8;      // top_k
constexpr int E_ = 64;     // experts
constexpr int H_ = 2048;   // hidden
constexpr int I_ = 768;    // moe intermediate
constexpr int C_ = (T_ * K_) / E_;   // 256 pairs per expert (balanced)
constexpr int TK_ = T_ * K_;         // 16384
constexpr int LDA = 80;              // LDS row stride (bf16): 160B -> 40-bank stride, 32B aligned

__device__ __forceinline__ bf16_t f2bf(float x) { return (bf16_t)x; }

// convert 16 fp32 (4 x float4) -> 16 bf16 into LDS
__device__ __forceinline__ void cvt_store16(bf16_t* d, const float4* r) {
#pragma unroll
    for (int i = 0; i < 4; ++i) {
        d[4*i+0] = f2bf(r[i].x); d[4*i+1] = f2bf(r[i].y);
        d[4*i+2] = f2bf(r[i].z); d[4*i+3] = f2bf(r[i].w);
    }
}

// assemble the wave32 bf16 A fragment (16x32): per lane two 16B chunks, K+0..7 and K+16..23
__device__ __forceinline__ v16bf load_afrag(const bf16_t* ap) {
    v8bf lo = *(const v8bf*)(ap);
    v8bf hi = *(const v8bf*)(ap + 16);
    v16bf a;
#pragma unroll
    for (int i = 0; i < 8; ++i) { a[i] = lo[i]; a[i+8] = hi[i]; }
    return a;
}

// ---------------------------------------------------------------------------
// Zero the output (harness poisons d_out; scatter-add needs zeros).
// ---------------------------------------------------------------------------
__global__ __launch_bounds__(256) void zero_kernel(float4* __restrict__ p, int n4) {
    int i = blockIdx.x * 256 + threadIdx.x;
    if (i < n4) p[i] = float4{0.f, 0.f, 0.f, 0.f};
}

// ---------------------------------------------------------------------------
// Routing: one wave32 per expert; ballot-compaction reproduces the stable
// argsort grouping of (token,slot) pairs by expert.
// ---------------------------------------------------------------------------
__global__ __launch_bounds__(32) void route_kernel(
    const int*   __restrict__ topk_idx,   // [T,K]
    const float* __restrict__ topk_w,     // [T,K]
    int*         __restrict__ pair_tok,   // [E,C]
    float*       __restrict__ pair_w)     // [E,C]
{
    const int e    = blockIdx.x;
    const int lane = threadIdx.x & 31;
    int count = 0;
    for (int base = 0; base < TK_; base += 32) {
        int i = base + lane;
        bool flag = (topk_idx[i] == e);
        unsigned mask = (unsigned)__ballot(flag);
        int pos = __popc(mask & ((1u << lane) - 1u));
        if (flag) {
            int slot = count + pos;
            if (slot < C_) {
                pair_tok[e * C_ + slot] = i / K_;
                pair_w  [e * C_ + slot] = topk_w[i];
            }
        }
        count += __popc(mask);
    }
}

// ---------------------------------------------------------------------------
// GEMM1 + fused SwiGLU. Block tile M=64 pairs, N = 32 gate cols + paired 32
// up cols. K staged 64-wide, double-buffered LDS: one barrier per 4 WMMAs
// per wave. fp32 weights converted to bf16 on the fly (read exactly once).
// ---------------------------------------------------------------------------
__global__ __launch_bounds__(256) void gemm1_kernel(
    const float* __restrict__ hidden,     // [T,H]
    const float* __restrict__ gup,        // [E,2I,H]
    const int*   __restrict__ pair_tok,   // [E,C]
    bf16_t*      __restrict__ h_buf)      // [E*C, I] bf16
{
    __shared__ bf16_t As[2][64][LDA];   // activations, rows = pair
    __shared__ bf16_t Bs[2][64][LDA];   // rows 0-31 gate, 32-63 up

    const int e   = blockIdx.z;
    const int n0  = blockIdx.x * 32;
    const int m0  = blockIdx.y * 64;
    const int tid  = threadIdx.x;
    const int lane = tid & 31;
    const int w    = tid >> 5;
    const int mi   = w & 3;
    const int ni   = w >> 2;

    // cooperative loaders: 64 rows x 64 k, 16 elems/thread
    const int lrow = tid >> 2;          // 0..63
    const int lcol = (tid & 3) * 16;    // 0,16,32,48

    const int   tok  = pair_tok[e * C_ + m0 + lrow];
    const float* arow = hidden + (size_t)tok * H_ + lcol;
    const int   wrow = (lrow < 32) ? (n0 + lrow) : (I_ + n0 + (lrow - 32));
    const float* brow = gup + ((size_t)e * (2 * I_) + wrow) * H_ + lcol;

    v8f accg = {};
    v8f accu = {};

    const int mloc = mi * 16 + (lane & 15);
    const int kA   = ((lane >> 4) & 1) * 8;
    const int nloc = ni * 16 + (lane & 15);
    const int kB   = ((lane >> 4) & 1) * 8;

    // prologue: stage 0 -> buffer 0
    {
        float4 ra[4], rb[4];
#pragma unroll
        for (int i = 0; i < 4; ++i) {
            ra[i] = *(const float4*)(arow + i * 4);
            rb[i] = *(const float4*)(brow + i * 4);
        }
        cvt_store16(&As[0][lrow][lcol], ra);
        cvt_store16(&Bs[0][lrow][lcol], rb);
        __syncthreads();
    }

    constexpr int NS = H_ / 64;   // 32 stages
    for (int s = 0; s < NS; ++s) {
        const int p = s & 1, q = p ^ 1;
        float4 ra[4], rb[4];
        if (s + 1 < NS) {
            const float* an = arow + (s + 1) * 64;
            const float* bn = brow + (s + 1) * 64;
#pragma unroll
            for (int i = 0; i < 4; ++i) {
                ra[i] = *(const float4*)(an + i * 4);
                rb[i] = *(const float4*)(bn + i * 4);
            }
        }
#pragma unroll
        for (int ks = 0; ks < 64; ks += 32) {
            v16bf a  = load_afrag(&As[p][mloc][ks + kA]);
            v16bf bg = load_afrag(&Bs[p][nloc][ks + kB]);
            v16bf bu = load_afrag(&Bs[p][32 + nloc][ks + kB]);
            accg = __builtin_amdgcn_wmma_f32_16x16x32_bf16(false, a, false, bg,
                                                           (short)0, accg, false, false);
            accu = __builtin_amdgcn_wmma_f32_16x16x32_bf16(false, a, false, bu,
                                                           (short)0, accu, false, false);
        }
        if (s + 1 < NS) {
            cvt_store16(&As[q][lrow][lcol], ra);
            cvt_store16(&Bs[q][lrow][lcol], rb);
        }
        __syncthreads();
    }

    // fused SwiGLU epilogue -> bf16 h
    const int mbase = m0 + mi * 16 + ((lane >> 4) & 1) * 8;
    const int ncol  = n0 + ni * 16 + (lane & 15);
#pragma unroll
    for (int r = 0; r < 8; ++r) {
        float g = accg[r], u = accu[r];
        float hv = (g / (1.0f + __expf(-g))) * u;
        h_buf[(size_t)(e * C_ + mbase + r) * I_ + ncol] = f2bf(hv);
    }
}

// ---------------------------------------------------------------------------
// GEMM2 + weighted scatter-add. A-tile (bf16 h) is DMA'd global->LDS with
// GLOBAL_LOAD_ASYNC_TO_LDS_B128 (ASYNCcnt), overlapping the WMMA stream;
// B-tile (fp32 down_proj) converts through VGPRs. Double-buffered, K-stage 64.
// ---------------------------------------------------------------------------
__global__ __launch_bounds__(256) void gemm2_kernel(
    const bf16_t* __restrict__ h_buf,     // [E*C, I] bf16
    const float*  __restrict__ down,      // [E,H,I]
    const int*    __restrict__ pair_tok,  // [E,C]
    const float*  __restrict__ pair_w,    // [E,C]
    float*        __restrict__ out)       // [T,H]
{
    __shared__ bf16_t As[2][64][LDA];
    __shared__ bf16_t Bs[2][32][LDA];

    const int e   = blockIdx.z;
    const int n0  = blockIdx.x * 32;
    const int m0  = blockIdx.y * 64;
    const int tid  = threadIdx.x;
    const int lane = tid & 31;
    const int w    = tid >> 5;
    const int mi   = w & 3;
    const int ni   = w >> 2;

    // A loader: 64 rows x 64 k bf16, 16 bf16 (32B) per thread, async DMA
    const int lrow = tid >> 2;
    const int lcol = (tid & 3) * 16;
    const bf16_t* arow = h_buf + (size_t)(e * C_ + m0 + lrow) * I_ + lcol;

    // B loader: 32 rows x 64 k fp32, 8 floats per thread
    const int brw = tid >> 3;           // 0..31
    const int bcl = (tid & 7) * 8;      // 0,8,...,56
    const float* brow = down + ((size_t)e * H_ + n0 + brw) * I_ + bcl;

    v8f acc = {};

    const int mloc = mi * 16 + (lane & 15);
    const int kA   = ((lane >> 4) & 1) * 8;
    const int nloc = ni * 16 + (lane & 15);
    const int kB   = ((lane >> 4) & 1) * 8;

    // issue one 32B async global->LDS copy per thread for stage kk into buf
    auto issue_async = [&](int buf, int kk) {
        unsigned lds = (unsigned)(unsigned long long)&As[buf][lrow][lcol];
        unsigned long long ga = (unsigned long long)(const void*)(arow + kk);
        asm volatile("global_load_async_to_lds_b128 %0, %1, off"
                     :: "v"(lds), "v"(ga) : "memory");
        asm volatile("global_load_async_to_lds_b128 %0, %1, off offset:16"
                     :: "v"(lds), "v"(ga) : "memory");
    };

    // prologue: stage 0 -> buffer 0
    {
        issue_async(0, 0);
        float4 rb[2];
        rb[0] = *(const float4*)(brow);
        rb[1] = *(const float4*)(brow + 4);
        bf16_t* dB = &Bs[0][brw][bcl];
        dB[0]=f2bf(rb[0].x); dB[1]=f2bf(rb[0].y); dB[2]=f2bf(rb[0].z); dB[3]=f2bf(rb[0].w);
        dB[4]=f2bf(rb[1].x); dB[5]=f2bf(rb[1].y); dB[6]=f2bf(rb[1].z); dB[7]=f2bf(rb[1].w);
        asm volatile("s_wait_asynccnt 0x0" ::: "memory");
        __syncthreads();
    }

    constexpr int NS = I_ / 64;   // 12 stages
    for (int s = 0; s < NS; ++s) {
        const int p = s & 1, q = p ^ 1;
        float4 rb[2];
        const bool more = (s + 1 < NS);
        if (more) {
            issue_async(q, (s + 1) * 64);
            const float* bn = brow + (s + 1) * 64;
            rb[0] = *(const float4*)(bn);
            rb[1] = *(const float4*)(bn + 4);
        }
#pragma unroll
        for (int ks = 0; ks < 64; ks += 32) {
            v16bf a  = load_afrag(&As[p][mloc][ks + kA]);
            v16bf bb = load_afrag(&Bs[p][nloc][ks + kB]);
            acc = __builtin_amdgcn_wmma_f32_16x16x32_bf16(false, a, false, bb,
                                                          (short)0, acc, false, false);
        }
        if (more) {
            bf16_t* dB = &Bs[q][brw][bcl];
            dB[0]=f2bf(rb[0].x); dB[1]=f2bf(rb[0].y); dB[2]=f2bf(rb[0].z); dB[3]=f2bf(rb[0].w);
            dB[4]=f2bf(rb[1].x); dB[5]=f2bf(rb[1].y); dB[6]=f2bf(rb[1].z); dB[7]=f2bf(rb[1].w);
            asm volatile("s_wait_asynccnt 0x0" ::: "memory");
        }
        __syncthreads();
    }

    // weighted scatter-add epilogue
    const int mbase = m0 + mi * 16 + ((lane >> 4) & 1) * 8;
    const int ncol  = n0 + ni * 16 + (lane & 15);
#pragma unroll
    for (int r = 0; r < 8; ++r) {
        int idx = e * C_ + mbase + r;
        float val = acc[r] * pair_w[idx];
        unsafeAtomicAdd(&out[(size_t)pair_tok[idx] * H_ + ncol], val);
    }
}

// ---------------------------------------------------------------------------
extern "C" void kernel_launch(void* const* d_in, const int* in_sizes, int n_in,
                              void* d_out, int out_size, void* d_ws, size_t ws_size,
                              hipStream_t stream) {
    const float* hidden   = (const float*)d_in[0];   // [T,H] fp32
    const int*   topk_idx = (const int*)  d_in[1];   // [T,K] int32
    const float* topk_w   = (const float*)d_in[2];   // [T,K] fp32
    const float* gup      = (const float*)d_in[3];   // [E,2I,H] fp32
    const float* down     = (const float*)d_in[4];   // [E,H,I] fp32
    float* out = (float*)d_out;                      // [T,H] fp32

    char* ws = (char*)d_ws;
    int*    pair_tok = (int*)ws;                            // 64 KB
    float*  pair_w   = (float*)(ws + (size_t)E_ * C_ * 4);  // 64 KB
    bf16_t* h_buf    = (bf16_t*)(ws + (size_t)E_ * C_ * 8); // ~24 MB

    int n4 = (T_ * H_) / 4;
    zero_kernel<<<(n4 + 255) / 256, 256, 0, stream>>>((float4*)out, n4);
    route_kernel<<<E_, 32, 0, stream>>>(topk_idx, topk_w, pair_tok, pair_w);
    gemm1_kernel<<<dim3(I_ / 32, C_ / 64, E_), 256, 0, stream>>>(
        hidden, gup, pair_tok, h_buf);
    gemm2_kernel<<<dim3(H_ / 32, C_ / 64, E_), 256, 0, stream>>>(
        h_buf, down, pair_tok, pair_w, out);
}
